// GNNStructEncoder_88510686036806
// MI455X (gfx1250) — matrix operand order repaired and checked by reading, not compiled, not run on hardware
//
#include <hip/hip_runtime.h>

typedef __attribute__((ext_vector_type(16))) _Float16 v16h;
typedef __attribute__((ext_vector_type(8)))  float    v8f;
typedef __attribute__((ext_vector_type(4)))  unsigned u32x4;
typedef __attribute__((ext_vector_type(4)))  int      i32x4;

#ifndef __has_builtin
#define __has_builtin(x) 0
#endif
#if __has_builtin(__builtin_amdgcn_global_load_async_to_lds_b128) && \
    __has_builtin(__builtin_amdgcn_s_wait_asynccnt)
#define USE_ASYNC_WSTAGE 1
#else
#define USE_ASYNC_WSTAGE 0
#endif

#define N_NODES 50000
#define N_EDGES 800000
#define D_IN    256
#define H_DIM   96
#define BN_EPS  1e-5f

// ---- workspace layout (bytes), 128B aligned ----
#define OFF_W0    0                      // 96*256 f16 = 49152
#define OFF_W1    49152                  // 96*96  f16 = 18432
#define OFF_W2    67584                  // 96*96  f16 = 18432
#define OFF_SUM   86016                  // 96 f32
#define OFF_SQ    86400                  // 96 f32
#define OFF_SCALE 86784                  // 96 f32
#define OFF_SHIFT 87168                  // 96 f32
#define OFF_H0    87552                  // 50000*96 f32 (reused for h)
#define OFF_Z     (87552 + 19200000)     // 50000*96 f32
// total = 38,487,552 bytes

struct Frag { union { v16h v; u32x4 q[2]; }; };

// Convert weights f32->f16 into ws; zero BN stat accumulators.
__global__ void k_prep(const float* __restrict__ W0, const float* __restrict__ W1,
                       const float* __restrict__ W2, char* __restrict__ ws) {
    _Float16* w0h = (_Float16*)(ws + OFF_W0);
    _Float16* w1h = (_Float16*)(ws + OFF_W1);
    _Float16* w2h = (_Float16*)(ws + OFF_W2);
    float*    st  = (float*)(ws + OFF_SUM);   // sum + sq contiguous = 192 floats
    const int t0 = H_DIM * D_IN, t1 = H_DIM * H_DIM, t2 = H_DIM * H_DIM;
    const int total = t0 + t1 + t2 + 192;
    for (int i = blockIdx.x * blockDim.x + threadIdx.x; i < total;
         i += gridDim.x * blockDim.x) {
        if (i < t0)                w0h[i] = (_Float16)W0[i];
        else if (i < t0 + t1)      w1h[i - t0] = (_Float16)W1[i - t0];
        else if (i < t0 + t1 + t2) w2h[i - t0 - t1] = (_Float16)W2[i - t0 - t1];
        else                       st[i - t0 - t1 - t2] = 0.f;
    }
}

// out[nrows,96] = op(A[nrows,K]) @ Bw^T + bias.   Bw: f16 [96,K] row-major.
// MODE 0: identity.  MODE 1: relu(A*scale[k] + shift[k]) (fused BatchNorm).
// One wave owns TWO 16-row tiles (M=32): each B fragment feeds 2 WMMAs,
// B fragments ping-pong between two register buffers. The last pair simply
// re-anchors to nrows-32 (16-aligned): the 16-row overlap with the previous
// wave stores byte-identical values, so there is NO tail path at all.
template <int MODE>
__global__ void k_gemm_wmma(const float* __restrict__ A, const _Float16* __restrict__ Bw,
                            const float* __restrict__ bias, float* __restrict__ out,
                            int nrows, int K,
                            const float* __restrict__ scale, const float* __restrict__ shift) {
    extern __shared__ char smem[];
    _Float16* ldsW = (_Float16*)smem;                       // [96][K] f16
    const int tid    = threadIdx.x;
    const int lane   = tid & 31;
    const int wave   = tid >> 5;
    const int nWaves = blockDim.x >> 5;
    const int nbytesW = H_DIM * K * 2;

    // ---- stage weights into LDS ----
#if USE_ASYNC_WSTAGE
    {   // CDNA5 async global->LDS DMA path (ASYNCcnt-tracked)
        char* bsrc  = (char*)Bw;                            // drop const
        char* ldstc = (char*)ldsW;
        for (int off = tid * 16; off < nbytesW; off += blockDim.x * 16) {
            __attribute__((address_space(1))) i32x4* g =
                (__attribute__((address_space(1))) i32x4*)(bsrc + off);
            __attribute__((address_space(3))) i32x4* l =
                (__attribute__((address_space(3))) i32x4*)(ldstc + off);
            __builtin_amdgcn_global_load_async_to_lds_b128(g, l, 0, 0);
        }
        __builtin_amdgcn_s_wait_asynccnt(0);
    }
#else
    {
        const unsigned* src = (const unsigned*)Bw;
        unsigned*       dst = (unsigned*)ldsW;
        const int nd = nbytesW >> 2;
        for (int i = tid; i < nd; i += blockDim.x) dst[i] = src[i];
    }
#endif
    __syncthreads();

    _Float16* ldsA = (_Float16*)(smem + (size_t)nbytesW + wave * 2048); // 32x32 f16

    const int ntiles = nrows >> 4;                          // 3125 (exact)
    const int pair   = blockIdx.x * nWaves + wave;
    if (pair * 2 >= ntiles) return;                         // wave-uniform exit

    const int r    = lane & 15;
    const int half = lane >> 4;
    long long m0 = (long long)pair * 32;
    if (m0 + 32 > (long long)nrows) m0 = (long long)nrows - 32; // re-anchor last pair

    const int colL = (lane & 15) << 1;                      // staging: float2/lane
    const int rowL = lane >> 4;
    const float* aBase = A + (m0 + rowL) * (long long)K + colL;
    _Float16*    lBase = ldsA + rowL * 32 + colL;

    v8f acc0[6] = {};
    v8f acc1[6] = {};

    for (int k0 = 0; k0 < K; k0 += 32) {
        float sc0 = 1.f, sh0 = 0.f, sc1 = 1.f, sh1 = 0.f;
        if (MODE == 1) {
            sc0 = scale[k0 + colL];     sh0 = shift[k0 + colL];
            sc1 = scale[k0 + colL + 1]; sh1 = shift[k0 + colL + 1];
        }
        // ---- stage 32x32 A tile as packed f16 dwords, 2 chunks of 8 rows ----
        // (chunking caps in-flight load registers at 16 VGPRs)
#pragma unroll
        for (int c = 0; c < 2; ++c) {
            float2 t[8];
#pragma unroll
            for (int j = 0; j < 8; ++j)
                t[j] = *(const float2*)(aBase + (c * 8 + j) * 2 * K); // b64, imm off
#pragma unroll
            for (int j = 0; j < 8; ++j) {
                float a0 = t[j].x, a1 = t[j].y;
                if (MODE == 1) {
                    a0 = fmaxf(fmaf(a0, sc0, sh0), 0.f);    // fused BN + ReLU
                    a1 = fmaxf(fmaf(a1, sc1, sh1), 0.f);
                }
                union { unsigned u; _Float16 h[2]; } p;
                p.h[0] = (_Float16)a0; p.h[1] = (_Float16)a1;
                *(unsigned*)(lBase + (c * 8 + j) * 64) = p.u; // ds_store_b32, imm off
            }
        }
        if (k0 + 32 < K)                                    // global_prefetch_b8
            __builtin_prefetch(aBase + 32, 0, 1);
        asm volatile("s_wait_dscnt 0" ::: "memory");        // CDNA5 split DS counter

        // ---- fragments (all ds_load_b128) ----
        Frag af0, af1, bfA, bfB;
        af0.q[0] = *(const u32x4*)&ldsA[r * 32 + (half << 3)];
        af0.q[1] = *(const u32x4*)&ldsA[r * 32 + 16 + (half << 3)];
        af1.q[0] = *(const u32x4*)&ldsA[(16 + r) * 32 + (half << 3)];
        af1.q[1] = *(const u32x4*)&ldsA[(16 + r) * 32 + 16 + (half << 3)];

        auto loadB = [&](Frag& b, int n) {
            const u32x4* bp = (const u32x4*)&ldsW[(n * 16 + r) * K + k0 + (half << 3)];
            b.q[0] = bp[0];
            b.q[1] = bp[2];
        };

        loadB(bfA, 0);
#pragma unroll
        for (int n = 0; n < 6; n += 2) {
            loadB(bfB, n + 1);
            acc0[n] = __builtin_amdgcn_wmma_f32_16x16x32_f16(
                false, af0.v, false, bfA.v, (short)0, acc0[n], false, false);
            acc1[n] = __builtin_amdgcn_wmma_f32_16x16x32_f16(
                false, af1.v, false, bfA.v, (short)0, acc1[n], false, false);
            if (n + 2 < 6) loadB(bfA, n + 2);
            acc0[n + 1] = __builtin_amdgcn_wmma_f32_16x16x32_f16(
                false, af0.v, false, bfB.v, (short)0, acc0[n + 1], false, false);
            acc1[n + 1] = __builtin_amdgcn_wmma_f32_16x16x32_f16(
                false, af1.v, false, bfB.v, (short)0, acc1[n + 1], false, false);
        }
        aBase += 32;                                        // advance K window
    }

    // ---- epilogue: acc[n][i] = C[m0 + i + 8*half][n*16 + r]; always valid ----
#pragma unroll
    for (int n = 0; n < 6; ++n) {
        const float bv = bias[n * 16 + r];
#pragma unroll
        for (int i = 0; i < 8; ++i)
            out[(m0 + i + (half << 3)) * H_DIM + n * 16 + r] = acc0[n][i] + bv;
    }
#pragma unroll
    for (int n = 0; n < 6; ++n) {
        const float bv = bias[n * 16 + r];
#pragma unroll
        for (int i = 0; i < 8; ++i)
            out[(m0 + 16 + i + (half << 3)) * H_DIM + n * 16 + r] = acc1[n][i] + bv;
    }
}

// z[dst[e]] += h0[src[e]]  (z pre-initialized to h0 => GIN eps=0 self term)
__global__ void k_scatter(const float* __restrict__ h0, const long long* __restrict__ ei,
                          float* __restrict__ z) {
    const long long total = (long long)N_EDGES * H_DIM;
    for (long long t = (long long)blockIdx.x * blockDim.x + threadIdx.x; t < total;
         t += (long long)gridDim.x * blockDim.x) {
        const int e = (int)(t / H_DIM);
        const int f = (int)(t - (long long)e * H_DIM);
        const long long s = ei[e];
        const long long d = ei[N_EDGES + e];
        atomicAdd(&z[d * H_DIM + f], h0[s * H_DIM + f]);   // L2-resident atomics
    }
}

// per-feature sum / sumsq (96 threads per block own one column each)
__global__ void k_bnstats(const float* __restrict__ h, float* __restrict__ sum,
                          float* __restrict__ sq) {
    const int f = threadIdx.x;                              // 0..95
    float s = 0.f, q = 0.f;
    for (int row = blockIdx.x; row < N_NODES; row += gridDim.x) {
        const float v = h[(long long)row * H_DIM + f];
        s += v; q += v * v;
    }
    atomicAdd(&sum[f], s);
    atomicAdd(&sq[f], q);
}

__global__ void k_bnfinish(const float* __restrict__ sum, const float* __restrict__ sq,
                           const float* __restrict__ gamma, const float* __restrict__ beta,
                           float* __restrict__ scale, float* __restrict__ shift) {
    const int f = threadIdx.x;
    if (f < H_DIM) {
        const float inv_n = 1.f / (float)N_NODES;
        const float mu  = sum[f] * inv_n;
        const float var = sq[f] * inv_n - mu * mu;          // biased (BN training)
        const float sc  = gamma[f] * rsqrtf(var + BN_EPS);
        scale[f] = sc;
        shift[f] = beta[f] - mu * sc;
    }
}

extern "C" void kernel_launch(void* const* d_in, const int* in_sizes, int n_in,
                              void* d_out, int out_size, void* d_ws, size_t ws_size,
                              hipStream_t stream) {
    const float*     x     = (const float*)d_in[0];
    const long long* ei    = (const long long*)d_in[1];   // int64 [2, E]
    const float*     W0    = (const float*)d_in[2];
    const float*     b0    = (const float*)d_in[3];
    const float*     W1    = (const float*)d_in[4];
    const float*     b1    = (const float*)d_in[5];
    const float*     gamma = (const float*)d_in[6];
    const float*     beta  = (const float*)d_in[7];
    const float*     W2    = (const float*)d_in[8];
    const float*     b2    = (const float*)d_in[9];
    char*  ws  = (char*)d_ws;
    float* out = (float*)d_out;

    _Float16* w0h = (_Float16*)(ws + OFF_W0);
    _Float16* w1h = (_Float16*)(ws + OFF_W1);
    _Float16* w2h = (_Float16*)(ws + OFF_W2);
    float* sum   = (float*)(ws + OFF_SUM);
    float* sq    = (float*)(ws + OFF_SQ);
    float* scale = (float*)(ws + OFF_SCALE);
    float* shift = (float*)(ws + OFF_SHIFT);
    float* h0    = (float*)(ws + OFF_H0);     // reused for h after the scatter
    float* z     = (float*)(ws + OFF_Z);

    k_prep<<<170, 256, 0, stream>>>(W0, W1, W2, ws);

    const int tiles  = N_NODES / 16;                       // 3125
    const int pairs  = (tiles + 1) / 2;                    // 1563 (M=32 per wave)
    const int WPB    = 6;                                  // waves per 192-thread block
    const int blocks = (pairs + WPB - 1) / WPB;            // 261
    const size_t smA = WPB * 2048;                         // per-wave 32x32 f16 staging
    const size_t sm0 = (size_t)H_DIM * D_IN * 2 + smA;     // 61,440 B
    const size_t sm1 = (size_t)H_DIM * H_DIM * 2 + smA;    // 30,720 B

    // h0 = x @ W0^T + b0
    k_gemm_wmma<0><<<blocks, 192, sm0, stream>>>(x, w0h, b0, h0, N_NODES, D_IN,
                                                 nullptr, nullptr);
    // z = h0 (self term), then scatter-add neighbors
    (void)hipMemcpyAsync(z, h0, (size_t)N_NODES * H_DIM * sizeof(float),
                         hipMemcpyDeviceToDevice, stream);
    k_scatter<<<40960, 256, 0, stream>>>(h0, ei, z);

    // h = z @ W1^T + b1   (overwrites h0 buffer)
    k_gemm_wmma<0><<<blocks, 192, sm1, stream>>>(z, w1h, b1, h0, N_NODES, H_DIM,
                                                 nullptr, nullptr);
    // BN training stats -> scale/shift
    k_bnstats<<<512, 96, 0, stream>>>(h0, sum, sq);
    k_bnfinish<<<1, 96, 0, stream>>>(sum, sq, gamma, beta, scale, shift);

    // out = relu(bn(h)) @ W2^T + b2   (BN+ReLU fused into A staging)
    k_gemm_wmma<1><<<blocks, 192, sm1, stream>>>(h0, w2h, b2, out, N_NODES, H_DIM,
                                                 scale, shift);
}
